// Head_55482387530041
// MI455X (gfx1250) — matrix operand order repaired and checked by reading, not compiled, not run on hardware
//
#include <hip/hip_runtime.h>

typedef __attribute__((ext_vector_type(16))) _Float16 v16h;
typedef __attribute__((ext_vector_type(8)))  _Float16 v8h;
typedef __attribute__((ext_vector_type(8)))  float    v8f;

#define WMMA_F16(A, B, Cacc) \
  __builtin_amdgcn_wmma_f32_16x16x32_f16(false, (A), false, (B), (short)0, (Cacc), false, false)

constexpr int BATCH = 8;
constexpr int T     = 2048;
constexpr int CDIM  = 768;
constexpr int H     = 64;
constexpr int BT    = BATCH * T;

// ---------------- Kernel 0: convert W [768][64] f32 -> Wt [64][768] f16 ----------------
__global__ void convert_w_kernel(const float* __restrict__ W, _Float16* __restrict__ Wt) {
  int idx = blockIdx.x * 256 + threadIdx.x;   // 0 .. 768*64-1
  int k = idx >> 6;                           // 0..767
  int n = idx & 63;                           // 0..63
  Wt[n * CDIM + k] = (_Float16)W[idx];
}

// ---------------- Kernel 1: q/k/v projections via WMMA ----------------
__global__ __launch_bounds__(128) void proj_qkv_kernel(
    const float* __restrict__ x,
    const _Float16* __restrict__ Wqt, const _Float16* __restrict__ Wkt,
    const _Float16* __restrict__ Wvt,
    _Float16* __restrict__ qh, _Float16* __restrict__ kh, _Float16* __restrict__ vt)
{
  const int lane  = threadIdx.x & 31;
  const int wave  = threadIdx.x >> 5;
  const int mtile = blockIdx.x * 4 + wave;   // 0..1023
  const int row0  = mtile * 16;
  const int rA    = lane & 15;
  const int hi    = lane >> 4;               // half-wave id
  const int kbA   = hi * 8;                  // A-layout K base
  const int koB   = hi * 16;                 // contiguous K base (A and B)

  const _Float16* const Wts[3] = {Wqt, Wkt, Wvt};
  const v8f zero = {};
  v8f acc[3][4];
#pragma unroll
  for (int w = 0; w < 3; ++w)
#pragma unroll
    for (int nt = 0; nt < 4; ++nt) acc[w][nt] = zero;

  for (int kc = 0; kc < CDIM / 32; ++kc) {
    const int k0 = kc * 32;
    const float* xr = x + (size_t)(row0 + rA) * CDIM + k0;
    float xs[16];
    *(float4*)(xs + 0)  = *(const float4*)(xr + koB);
    *(float4*)(xs + 4)  = *(const float4*)(xr + koB + 4);
    *(float4*)(xs + 8)  = *(const float4*)(xr + koB + 8);
    *(float4*)(xs + 12) = *(const float4*)(xr + koB + 12);
    v16h a;
#pragma unroll
    for (int e = 0; e < 16; ++e) a[e] = (_Float16)xs[e];

#pragma unroll
    for (int w = 0; w < 3; ++w) {
#pragma unroll
      for (int nt = 0; nt < 4; ++nt) {
        const v16h b = *(const v16h*)(Wts[w] + (size_t)(nt * 16 + rA) * CDIM + k0 + koB);
        acc[w][nt] = WMMA_F16(a, b, acc[w][nt]);
      }
    }
  }

  const int bb = row0 >> 11;               // batch
  const int t0 = (row0 & (T - 1)) + 8 * hi;
#pragma unroll
  for (int nt = 0; nt < 4; ++nt) {
#pragma unroll
    for (int r = 0; r < 8; ++r) {
      const size_t off = (size_t)(row0 + r + 8 * hi) * H + nt * 16 + rA;
      qh[off] = (_Float16)acc[0][nt][r];
      kh[off] = (_Float16)acc[1][nt][r];
    }
    v8h pv;
#pragma unroll
    for (int r = 0; r < 8; ++r) pv[r] = (_Float16)acc[2][nt][r];
    *(v8h*)(vt + ((size_t)(bb * H + nt * 16 + rA)) * T + t0) = pv;
  }
}

// ---------------- Kernel 2: fused flash attention with async LDS staging ----------------
// 4 waves/WG share each 32-key K/V tile, staged by GLOBAL_LOAD_ASYNC_TO_LDS (double-buffered).
// Each wave owns 16 query rows; online softmax entirely in registers.

// Stage one 32-key tile: K tile = 4 KB contiguous from kh; V tile = 64 rows x 64 B from vt.
// 128 threads x 4 async b128 ops. INST_OFFSET applies to both LDS and global addresses.
__device__ __forceinline__ void stage_kv_async(const _Float16* kb, const _Float16* vb,
                                               int key0, _Float16* kbuf, _Float16* vbuf,
                                               int tid) {
  {
    unsigned int  l0 = (unsigned int)(uintptr_t)kbuf + (unsigned int)tid * 16u;
    unsigned long long g0 = (unsigned long long)(uintptr_t)(kb + (size_t)key0 * H)
                            + (unsigned long long)tid * 16ull;
    asm volatile("global_load_async_to_lds_b128 %0, %1, off"
                 :: "v"(l0), "v"(g0) : "memory");
    asm volatile("global_load_async_to_lds_b128 %0, %1, off offset:2048"
                 :: "v"(l0), "v"(g0) : "memory");
  }
  {
    const int h = tid >> 1;                 // 0..63
    const int c = (tid & 1) * 32;           // byte chunk within 64B row
    unsigned int  l1 = (unsigned int)(uintptr_t)vbuf + (unsigned int)(h * 64 + c);
    unsigned long long g1 = (unsigned long long)(uintptr_t)(vb + (size_t)h * T + key0)
                            + (unsigned long long)c;
    asm volatile("global_load_async_to_lds_b128 %0, %1, off"
                 :: "v"(l1), "v"(g1) : "memory");
    asm volatile("global_load_async_to_lds_b128 %0, %1, off offset:16"
                 :: "v"(l1), "v"(g1) : "memory");
  }
}

__global__ __launch_bounds__(128) void attn_kernel(
    const _Float16* __restrict__ qh, const _Float16* __restrict__ kh,
    const _Float16* __restrict__ vt, float* __restrict__ out)
{
  __shared__ __align__(32) _Float16 kbuf[2][32 * H];    // [buf][key][h]   4 KB each
  __shared__ __align__(32) _Float16 vbuf[2][H * 32];    // [buf][h][key]   4 KB each
  __shared__ __align__(32) _Float16 pstage[4][16 * 32]; // per-wave P relayout

  const int tid   = threadIdx.x;
  const int lane  = tid & 31;
  const int wave  = tid >> 5;
  const int mtile = blockIdx.x * 4 + wave;   // 4 tiles of one batch per WG
  const int row0  = mtile * 16;
  const int bb    = row0 >> 11;
  const int rA    = lane & 15;
  const int hi    = lane >> 4;
  const int kbA   = hi * 8;
  const int koB   = hi * 16;

  const _Float16* kb = kh + (size_t)bb * T * H;   // [T][H]
  const _Float16* vb = vt + (size_t)bb * H * T;   // [H][T]

  // q A-operands for K=0..31 and K=32..63
  const _Float16* qr = qh + (size_t)(row0 + rA) * H;
  v8h q0 = *(const v8h*)(qr + koB);
  v8h q1 = *(const v8h*)(qr + koB + 8);
  v8h q2 = *(const v8h*)(qr + 32 + koB);
  v8h q3 = *(const v8h*)(qr + 32 + koB + 8);
  const v16h aq0 = __builtin_shufflevector(q0, q1, 0,1,2,3,4,5,6,7,8,9,10,11,12,13,14,15);
  const v16h aq1 = __builtin_shufflevector(q2, q3, 0,1,2,3,4,5,6,7,8,9,10,11,12,13,14,15);

  const v8f zero = {};
  float m[8], l[8];
  v8f o[4];
#pragma unroll
  for (int r = 0; r < 8; ++r) { m[r] = -1e30f; l[r] = 0.0f; }
#pragma unroll
  for (int nt = 0; nt < 4; ++nt) o[nt] = zero;

  _Float16* pw = &pstage[wave][0];
  const float sc = 0.125f;   // 1/sqrt(64)

  // prime the pipeline: tile 0 -> buffer 0
  stage_kv_async(kb, vb, 0, &kbuf[0][0], &vbuf[0][0], tid);

  for (int kt = 0; kt < T / 32; ++kt) {
    __syncthreads();   // everyone done reading the buffer we are about to overwrite
    if (kt + 1 < T / 32) {
      stage_kv_async(kb, vb, (kt + 1) * 32, &kbuf[(kt + 1) & 1][0], &vbuf[(kt + 1) & 1][0], tid);
      asm volatile("s_wait_asynccnt 0x4" ::: "memory");  // retire current tile's 4 ops (in-order)
    } else {
      asm volatile("s_wait_asynccnt 0x0" ::: "memory");
    }
    __syncthreads();   // current tile visible to all waves

    const _Float16* kcur = &kbuf[kt & 1][0];
    const _Float16* vcur = &vbuf[kt & 1][0];

    // ---- S = q @ k^T for 2 x 16-key tiles (K=64 split into two WMMAs) ----
    v8f s0 = zero, s1 = zero;
    {
      const _Float16* kr0 = kcur + (size_t)rA * H;
      const v16h bk0 = *(const v16h*)(kr0 + koB);
      const v16h bk1 = *(const v16h*)(kr0 + 32 + koB);
      s0 = WMMA_F16(aq0, bk0, s0);
      s0 = WMMA_F16(aq1, bk1, s0);
      const _Float16* kr1 = kcur + (size_t)(16 + rA) * H;
      const v16h bk2 = *(const v16h*)(kr1 + koB);
      const v16h bk3 = *(const v16h*)(kr1 + 32 + koB);
      s1 = WMMA_F16(aq0, bk2, s1);
      s1 = WMMA_F16(aq1, bk3, s1);
    }
    // ---- online softmax (row M = r + 8*hi lives across one 16-lane half) ----
#pragma unroll
    for (int r = 0; r < 8; ++r) {
      float a0 = s0[r] * sc, a1 = s1[r] * sc;
      float mx = fmaxf(a0, a1);
      mx = fmaxf(mx, __shfl_xor(mx, 1, 32));
      mx = fmaxf(mx, __shfl_xor(mx, 2, 32));
      mx = fmaxf(mx, __shfl_xor(mx, 4, 32));
      mx = fmaxf(mx, __shfl_xor(mx, 8, 32));
      const float mn   = fmaxf(m[r], mx);
      const float corr = __expf(m[r] - mn);
      const float p0   = __expf(a0 - mn);
      const float p1   = __expf(a1 - mn);
      float ps = p0 + p1;
      ps += __shfl_xor(ps, 1, 32);
      ps += __shfl_xor(ps, 2, 32);
      ps += __shfl_xor(ps, 4, 32);
      ps += __shfl_xor(ps, 8, 32);
      l[r] = l[r] * corr + ps;
      m[r] = mn;
#pragma unroll
      for (int nt = 0; nt < 4; ++nt) o[nt][r] *= corr;
      pw[(r + 8 * hi) * 32 + rA]      = (_Float16)p0;
      pw[(r + 8 * hi) * 32 + 16 + rA] = (_Float16)p1;
    }
    asm volatile("s_wait_dscnt 0" ::: "memory");   // same-wave DS ops are in-order
    v8h pa0 = *(const v8h*)(pw + rA * 32 + koB);
    v8h pa1 = *(const v8h*)(pw + rA * 32 + koB + 8);
    const v16h ap = __builtin_shufflevector(pa0, pa1, 0,1,2,3,4,5,6,7,8,9,10,11,12,13,14,15);
    // ---- O += P @ V ; vbuf is [h][key] so B-operand is key-contiguous ----
#pragma unroll
    for (int nt = 0; nt < 4; ++nt) {
      const v16h bv = *(const v16h*)(vcur + (size_t)(nt * 16 + rA) * 32 + koB);
      o[nt] = WMMA_F16(ap, bv, o[nt]);
    }
  }

  // ---- normalize and store [B][T][H] fp32 ----
#pragma unroll
  for (int r = 0; r < 8; ++r) {
    const float inv = 1.0f / l[r];
#pragma unroll
    for (int nt = 0; nt < 4; ++nt)
      out[(size_t)(row0 + r + 8 * hi) * H + nt * 16 + rA] = o[nt][r] * inv;
  }
}

extern "C" void kernel_launch(void* const* d_in, const int* in_sizes, int n_in,
                              void* d_out, int out_size, void* d_ws, size_t ws_size,
                              hipStream_t stream) {
  (void)in_sizes; (void)n_in; (void)out_size; (void)ws_size;
  const float* x  = (const float*)d_in[0];
  const float* Wk = (const float*)d_in[1];
  const float* Wq = (const float*)d_in[2];
  const float* Wv = (const float*)d_in[3];
  float* out = (float*)d_out;

  _Float16* ws  = (_Float16*)d_ws;
  _Float16* qh  = ws;                                  // BT*H halves
  _Float16* kh  = qh + (size_t)BT * H;
  _Float16* vt  = kh + (size_t)BT * H;
  _Float16* wqt = vt + (size_t)BT * H;                 // 768*64 halves each
  _Float16* wkt = wqt + (size_t)CDIM * H;
  _Float16* wvt = wkt + (size_t)CDIM * H;

  convert_w_kernel<<<(CDIM * H) / 256, 256, 0, stream>>>(Wq, wqt);
  convert_w_kernel<<<(CDIM * H) / 256, 256, 0, stream>>>(Wk, wkt);
  convert_w_kernel<<<(CDIM * H) / 256, 256, 0, stream>>>(Wv, wvt);
  proj_qkv_kernel<<<BT / 64, 128, 0, stream>>>(x, wqt, wkt, wvt, qh, kh, vt);
  attn_kernel<<<BT / 64, 128, 0, stream>>>(qh, kh, vt, out);
}
